// LSTMModel_19748259627756
// MI455X (gfx1250) — matrix operand order repaired and checked by reading, not compiled, not run on hardware
//
#include <hip/hip_runtime.h>

// Problem constants (from reference): B=128, T=512, V=256, H=512, L=2
#define BB 128
#define TT 512
#define VV 256
#define HH 512

typedef __attribute__((ext_vector_type(16))) __bf16 v16bf;
typedef __attribute__((ext_vector_type(8)))  __bf16 v8bf;
typedef __attribute__((ext_vector_type(8)))  float  v8f;
typedef int v4i __attribute__((vector_size(16)));   // matches builtin signature

#define WMMA_BF16(a, b, c)                                                    \
  __builtin_amdgcn_wmma_f32_16x16x32_bf16(false, (a), false, (b), (short)0,   \
                                          (c), false, false)

// ---- CDNA5 async global->LDS path (ASYNCcnt), with safe fallback ----------
#define GLOBAL_AS __attribute__((address_space(1)))
#define LDS_AS    __attribute__((address_space(3)))

#if defined(__HIP_DEVICE_COMPILE__) &&                                        \
    __has_builtin(__builtin_amdgcn_global_load_async_to_lds_b128) &&          \
    __has_builtin(__builtin_amdgcn_s_wait_asynccnt)
#define HAVE_ASYNC_LDS 1
#define ASYNC_WAIT() __builtin_amdgcn_s_wait_asynccnt(0)
#else
#define HAVE_ASYNC_LDS 0
#define ASYNC_WAIT()
#endif

static __device__ __forceinline__ v16bf cat8(v8bf lo, v8bf hi) {
  return __builtin_shufflevector(lo, hi, 0, 1, 2, 3, 4, 5, 6, 7, 8, 9, 10, 11,
                                 12, 13, 14, 15);
}

// A-operand (16-bit, 16x32): lane<16 -> row m0+lane, K = {k0..k0+7, k0+16..k0+23}
//                            lane>=16 -> row m0+lane-16, K shifted by +8.
static __device__ __forceinline__ v16bf load_a_bf(const __bf16* base,
                                                  int stride, int m0, int k0,
                                                  int lane) {
  const __bf16* p =
      base + (size_t)(m0 + (lane & 15)) * stride + k0 + (lane >> 4) * 8;
  v8bf lo = *(const v8bf*)p;
  v8bf hi = *(const v8bf*)(p + 16);
  return cat8(lo, hi);
}

// Same A layout, but source is fp32 (x tensor) converted in-register to bf16.
static __device__ __forceinline__ v16bf load_a_f32(const float* base,
                                                   int stride, int m0, int k0,
                                                   int lane) {
  const float* p =
      base + (size_t)(m0 + (lane & 15)) * stride + k0 + (lane >> 4) * 8;
  v8f f0 = *(const v8f*)p;
  v8f f1 = *(const v8f*)(p + 16);
  v16bf a;
#pragma unroll
  for (int i = 0; i < 8; ++i) {
    a[i] = (__bf16)f0[i];
    a[8 + i] = (__bf16)f1[i];
  }
  return a;
}

// B-operand read from an LDS-staged chunk Bs[64 rows][32 K] (rows = 4 gates x 16
// cols). Lane<16 -> col (lane), K = {0..7, 16..23} ; lane>=16 -> K shifted by +8.
static __device__ __forceinline__ v16bf load_b_lds(const __bf16* bs, int gate,
                                                   int lane) {
  const __bf16* p = bs + (gate * 16 + (lane & 15)) * 32 + (lane >> 4) * 8;
  v8bf lo = *(const v8bf*)p;
  v8bf hi = *(const v8bf*)(p + 16);
  return cat8(lo, hi);
}

// Direct-from-global B (used by the small FC kernel only).
static __device__ __forceinline__ v16bf load_b_w(const __bf16* w, int K,
                                                 int n0, int k0, int lane) {
  const __bf16* p = w + (size_t)(n0 + (lane & 15)) * K + k0 + (lane >> 4) * 8;
  v8bf lo = *(const v8bf*)p;
  v8bf hi = *(const v8bf*)(p + 16);
  return cat8(lo, hi);
}

// Stage one 64x32 weight chunk (4 gates x 16 cols x 32 K, 4 KB) into LDS.
// 256 threads: thread -> (row = tid>>2, 16-byte segment = tid&3). Async DMA on
// CDNA5 (GLOBAL_LOAD_ASYNC_TO_LDS_B128, ASYNCcnt); plain ld/st fallback.
static __device__ __forceinline__ void stage_b_chunk(
    const __bf16* __restrict__ w, int K, int nb, int k0,
    __bf16* __restrict__ bs, int tid) {
  const int rr = tid >> 2;    // 0..63
  const int seg = tid & 3;    // 4 x 8 bf16 = 32 K values per row
  const int gate = rr >> 4;
  const int r = rr & 15;
  const __bf16* src = w + (size_t)(gate * HH + nb + r) * K + k0 + seg * 8;
  __bf16* dst = bs + rr * 32 + seg * 8;
#if HAVE_ASYNC_LDS
  __builtin_amdgcn_global_load_async_to_lds_b128((GLOBAL_AS v4i*)src,
                                                 (LDS_AS v4i*)dst, 0, 0);
#else
  *(v8bf*)dst = *(const v8bf*)src;
#endif
}

static __device__ __forceinline__ float sigmoidf_(float x) {
  return 1.0f / (1.0f + __expf(-x));
}

// One LSTM timestep for one layer:
//   gates[B,4H] = x_t[B,KIN] @ wih^T + h_prev[B,H] @ whh^T + bias
// Block = 8 waves = all 8 batch tiles for ONE 16-col hidden group, so the
// 64-row weight slab is shared by the whole block: it is streamed through a
// double-buffered LDS pipeline (async DMA overlapped with WMMA), removing the
// 8x redundant L2 reads. Each wave carries all 4 gate accumulators so the
// i/f/g/o recombination is wave-local.
template <bool XF32, int KIN>
__global__ void __launch_bounds__(256) lstm_step(
    const float* __restrict__ xf, const __bf16* __restrict__ xb, int t,
    const __bf16* __restrict__ wih, const __bf16* __restrict__ whh,
    const float* __restrict__ bias, const __bf16* __restrict__ h_in,
    __bf16* __restrict__ h_out, float* __restrict__ c, float* __restrict__ hN) {
  __shared__ __bf16 bs[2][64 * 32];  // 2 x 4 KB double buffer

  const int tid = threadIdx.x;
  const int lane = tid & 31;
  const int m0 = (tid >> 5) * 16;     // batch-tile base (wave id)
  const int nb = blockIdx.x * 16;     // hidden-column base (block-uniform)

  constexpr int NIN = KIN / 32;       // input-GEMM chunks
  constexpr int NC = NIN + HH / 32;   // + recurrent chunks

  auto issue = [&](int cc) {
    if (cc < NIN) stage_b_chunk(wih, KIN, nb, cc * 32, bs[cc & 1], tid);
    else          stage_b_chunk(whh, HH, nb, (cc - NIN) * 32, bs[cc & 1], tid);
  };

  v8f ai = {}, af = {}, ag = {}, ao = {};
  issue(0);
  for (int cc = 0; cc < NC; ++cc) {
    ASYNC_WAIT();        // this thread's DMA for chunk cc has landed in LDS
    __syncthreads();     // ...and everyone else's; also retires chunk cc-1 reads
    if (cc + 1 < NC) issue(cc + 1);  // prefetch next chunk into other buffer

    v16bf a;
    if (cc < NIN) {
      if constexpr (XF32)
        a = load_a_f32(xf + (size_t)t * VV, TT * VV, m0, cc * 32, lane);
      else
        a = load_a_bf(xb, KIN, m0, cc * 32, lane);
    } else {
      a = load_a_bf(h_in, HH, m0, (cc - NIN) * 32, lane);
    }
    const __bf16* B = bs[cc & 1];
    ai = WMMA_BF16(a, load_b_lds(B, 0, lane), ai);
    af = WMMA_BF16(a, load_b_lds(B, 1, lane), af);
    ag = WMMA_BF16(a, load_b_lds(B, 2, lane), ag);
    ao = WMMA_BF16(a, load_b_lds(B, 3, lane), ao);
  }

  // C/D layout: reg r, lanes 0-15 -> (M=r, N=lane); lanes 16-31 -> (M=r+8).
  const int col = nb + (lane & 15);
  const int radd = (lane >> 4) * 8;
  const float bi = bias[0 * HH + col];
  const float bf_ = bias[1 * HH + col];
  const float bg = bias[2 * HH + col];
  const float bo = bias[3 * HH + col];
#pragma unroll
  for (int r = 0; r < 8; ++r) {
    const int b = m0 + r + radd;
    const int idx = b * HH + col;
    const float gi = sigmoidf_(ai[r] + bi);
    const float gf = sigmoidf_(af[r] + bf_);
    const float gg = tanhf(ag[r] + bg);
    const float go = sigmoidf_(ao[r] + bo);
    const float cn = gf * c[idx] + gi * gg;
    c[idx] = cn;                       // lives in d_out c_n region
    const float hn = go * tanhf(cn);
    hN[idx] = hn;                      // lives in d_out h_n region
    h_out[idx] = (__bf16)hn;           // bf16 ping-pong for next step's A
  }
}

// logits[128,256] = h_T @ fc_w^T + fc_b   (M=128, N=256, K=512; one-shot, tiny)
__global__ void __launch_bounds__(256) fc_kernel(const __bf16* __restrict__ h,
                                                 const __bf16* __restrict__ w,
                                                 const float* __restrict__ b,
                                                 float* __restrict__ out) {
  const int lane = threadIdx.x & 31;
  const int gw = blockIdx.x * 8 + (threadIdx.x >> 5);
  const int m0 = (gw & 7) * 16;
  const int n0 = (gw >> 3) * 16;
  v8f acc = {};
#pragma unroll 4
  for (int k0 = 0; k0 < HH; k0 += 32) {
    v16bf a = load_a_bf(h, HH, m0, k0, lane);
    v16bf bb = load_b_w(w, HH, n0, k0, lane);
    acc = WMMA_BF16(a, bb, acc);
  }
  const int col = n0 + (lane & 15);
  const int radd = (lane >> 4) * 8;
  const float bc = b[col];
#pragma unroll
  for (int r = 0; r < 8; ++r) out[(m0 + r + radd) * VV + col] = acc[r] + bc;
}

__global__ void cvt_bf16(const float* __restrict__ in, __bf16* __restrict__ out,
                         int n) {
  int i = blockIdx.x * blockDim.x + threadIdx.x;
  if (i < n) out[i] = (__bf16)in[i];
}

__global__ void add_bias(const float* __restrict__ a,
                         const float* __restrict__ b, float* __restrict__ o,
                         int n) {
  int i = blockIdx.x * blockDim.x + threadIdx.x;
  if (i < n) o[i] = a[i] + b[i];
}

__global__ void zero_u32(unsigned* __restrict__ p, int n) {
  int i = blockIdx.x * blockDim.x + threadIdx.x;
  if (i < n) p[i] = 0u;
}

extern "C" void kernel_launch(void* const* d_in, const int* in_sizes, int n_in,
                              void* d_out, int out_size, void* d_ws,
                              size_t ws_size, hipStream_t stream) {
  (void)in_sizes; (void)n_in; (void)out_size; (void)ws_size;
  const float* x    = (const float*)d_in[0];   // [B,T,V]
  const float* wih0 = (const float*)d_in[1];   // [4H,V]
  const float* whh0 = (const float*)d_in[2];   // [4H,H]
  const float* bih0 = (const float*)d_in[3];
  const float* bhh0 = (const float*)d_in[4];
  const float* wih1 = (const float*)d_in[5];   // [4H,H]
  const float* whh1 = (const float*)d_in[6];   // [4H,H]
  const float* bih1 = (const float*)d_in[7];
  const float* bhh1 = (const float*)d_in[8];
  const float* fcw  = (const float*)d_in[9];   // [V,H]
  const float* fcb  = (const float*)d_in[10];  // [V]

  // Workspace carve (~8.1 MB total)
  char* ws = (char*)d_ws;
  size_t off = 0;
  auto carve = [&](size_t bytes) -> char* {
    char* p = ws + off;
    off += (bytes + 255) & ~(size_t)255;
    return p;
  };
  __bf16* wih0b = (__bf16*)carve((size_t)4 * HH * VV * 2);
  __bf16* whh0b = (__bf16*)carve((size_t)4 * HH * HH * 2);
  __bf16* wih1b = (__bf16*)carve((size_t)4 * HH * HH * 2);
  __bf16* whh1b = (__bf16*)carve((size_t)4 * HH * HH * 2);
  __bf16* fcwb  = (__bf16*)carve((size_t)VV * HH * 2);
  float*  bias0 = (float*)carve((size_t)4 * HH * 4);
  float*  bias1 = (float*)carve((size_t)4 * HH * 4);
  __bf16* h0b[2], *h1b[2];
  h0b[0] = (__bf16*)carve((size_t)BB * HH * 2);
  h0b[1] = (__bf16*)carve((size_t)BB * HH * 2);
  h1b[0] = (__bf16*)carve((size_t)BB * HH * 2);
  h1b[1] = (__bf16*)carve((size_t)BB * HH * 2);

  // d_out partitions: logits[B,V] | h_n[2,B,H] | c_n[2,B,H]
  float* out    = (float*)d_out;
  float* logits = out;
  float* hN0 = out + BB * VV;
  float* hN1 = hN0 + BB * HH;
  float* c0  = hN1 + BB * HH;
  float* c1  = c0 + BB * HH;

  auto grid1d = [](int n) { return dim3((unsigned)((n + 255) / 256)); };

  // One-time prep: weights -> bf16, bias sums, zero state (re-done every call
  // so replays are deterministic; no reliance on prior buffer contents).
  cvt_bf16<<<grid1d(4 * HH * VV), 256, 0, stream>>>(wih0, wih0b, 4 * HH * VV);
  cvt_bf16<<<grid1d(4 * HH * HH), 256, 0, stream>>>(whh0, whh0b, 4 * HH * HH);
  cvt_bf16<<<grid1d(4 * HH * HH), 256, 0, stream>>>(wih1, wih1b, 4 * HH * HH);
  cvt_bf16<<<grid1d(4 * HH * HH), 256, 0, stream>>>(whh1, whh1b, 4 * HH * HH);
  cvt_bf16<<<grid1d(VV * HH), 256, 0, stream>>>(fcw, fcwb, VV * HH);
  add_bias<<<grid1d(4 * HH), 256, 0, stream>>>(bih0, bhh0, bias0, 4 * HH);
  add_bias<<<grid1d(4 * HH), 256, 0, stream>>>(bih1, bhh1, bias1, 4 * HH);
  const int nHu32_bf = BB * HH * 2 / 4;  // bf16 h buffer in u32 units
  zero_u32<<<grid1d(nHu32_bf), 256, 0, stream>>>((unsigned*)h0b[0], nHu32_bf);
  zero_u32<<<grid1d(nHu32_bf), 256, 0, stream>>>((unsigned*)h0b[1], nHu32_bf);
  zero_u32<<<grid1d(nHu32_bf), 256, 0, stream>>>((unsigned*)h1b[0], nHu32_bf);
  zero_u32<<<grid1d(nHu32_bf), 256, 0, stream>>>((unsigned*)h1b[1], nHu32_bf);
  zero_u32<<<grid1d(BB * HH), 256, 0, stream>>>((unsigned*)c0, BB * HH);
  zero_u32<<<grid1d(BB * HH), 256, 0, stream>>>((unsigned*)c1, BB * HH);

  // Sequential scan: stream ordering carries the timestep dependency.
  // h ping-pongs so no wave reads a buffer another wave of the same launch
  // is writing. Layer-1 step t consumes layer-0's freshly written h.
  int p = 0;
  for (int t = 0; t < TT; ++t) {
    lstm_step<true, VV><<<32, 256, 0, stream>>>(
        x, nullptr, t, wih0b, whh0b, bias0, h0b[p], h0b[1 - p], c0, hN0);
    lstm_step<false, HH><<<32, 256, 0, stream>>>(
        nullptr, h0b[1 - p], 0, wih1b, whh1b, bias1, h1b[p], h1b[1 - p], c1,
        hN1);
    p ^= 1;
  }

  // logits from final layer-1 hidden state (h1b[p] = last written buffer).
  fc_kernel<<<16, 256, 0, stream>>>(h1b[p], fcwb, fcb, logits);
}
